// LightGlueSelfAttentionBlock_65515431133857
// MI455X (gfx1250) — matrix operand order repaired and checked by reading, not compiled, not run on hardware
//
#include <hip/hip_runtime.h>

// ---------------------------------------------------------------------------
// LightGlue self-attention block for MI455X (gfx1250), bf16 WMMA path with
// async global->LDS tile staging (ASYNCcnt pipeline) and flash attention.
// B=4, N=4096, D=256, H=4, HD=64. wave32 only.
// ---------------------------------------------------------------------------

#define B_DIM 4
#define N_DIM 4096
#define D_DIM 256
#define H_DIM 4
#define HD_DIM 64
#define M_ROWS (B_DIM * N_DIM)       // 16384
#define LN_EPS 1e-5f

typedef __attribute__((ext_vector_type(16))) __bf16 v16bf;
typedef __attribute__((ext_vector_type(8)))  float  v8f;

union Frag16 {            // one 16x32 (A) or 32x16 (B) bf16 WMMA fragment
    v16bf v;
    uint4 q[2];
};

static __device__ __forceinline__ v8f wmma_bf16(const v16bf a, const v16bf b, v8f c) {
    // (neg_a, A, neg_b, B, c_mod, C, reuse_a, reuse_b)
    return __builtin_amdgcn_wmma_f32_16x16x32_bf16(false, a, false, b, (short)0, c, false, false);
}

// CDNA5 async DMA: 16 bytes global -> LDS, tracked by ASYNCcnt (no VGPR data).
static __device__ __forceinline__ void async_b128(const __bf16* g, __bf16* l) {
    unsigned lds_off = (unsigned)(size_t)l;   // low 32 bits of flat LDS addr = DS offset
    asm volatile("global_load_async_to_lds_b128 %0, %1, off"
                 :: "v"(lds_off), "v"(g) : "memory");
}
static __device__ __forceinline__ void wait_async0() {
    asm volatile("s_wait_asynccnt 0x0" ::: "memory");
}

// ---------------------------------------------------------------------------
// fp32 -> bf16 conversion kernels
// ---------------------------------------------------------------------------
__global__ void cvt_f32_bf16(const float* __restrict__ src, __bf16* __restrict__ dst, int n) {
    int i = blockIdx.x * 256 + threadIdx.x;
    if (i < n) dst[i] = (__bf16)src[i];
}

// descriptors [16384,256] f32 -> first half of concat buffer [16384,512] bf16
__global__ void desc_to_hcat(const float* __restrict__ d, __bf16* __restrict__ hcat) {
    int i = blockIdx.x * 256 + threadIdx.x;          // 16384*256 threads
    int row = i >> 8, c = i & 255;
    hcat[(size_t)row * 512 + c] = (__bf16)d[i];
}

// ---------------------------------------------------------------------------
// Generic bf16 GEMM:  out[M, Nout] = A[M,K](bf16) @ W[Nout,K](bf16)^T + bias
// Workgroup = 128 threads (4 waves); WG tile = 128x64; wave tile = 32x64.
// W tile (64x32) double-buffered in LDS via async loads; 8 WMMA per K-step.
// Epilogue: bf16 out (outB) or f32 out (outF), optional f32 residual add.
// ---------------------------------------------------------------------------
#define WS_STRIDE 40   // 32+8 bf16 elems -> 80B rows: 16B aligned, conflict-free
#define WS_TILE   (64 * WS_STRIDE)

__global__ __launch_bounds__(128) void gemm_wmma(
    const __bf16* __restrict__ A, int lda,
    const __bf16* __restrict__ W, int ldw,
    const float*  __restrict__ bias,
    const float*  __restrict__ resid, int resid_ld,
    float* __restrict__ outF, __bf16* __restrict__ outB, int ldo,
    int K)
{
    __shared__ __align__(16) __bf16 Ws[2 * WS_TILE];

    const int lane  = threadIdx.x & 31;
    const int w     = threadIdx.x >> 5;
    const int nbase = blockIdx.x * 64;
    const int mbase = blockIdx.y * 128 + w * 32;
    const int nlow  = lane & 15;
    const int hi    = lane >> 4;
    const int kA0   = hi * 16;    // A-frag K sub-offset per lane half
    const int kB0   = hi * 16;    // B-frag K sub-offset per lane half

    v8f acc[2][4] = {};

    auto stage_W = [&](int k0, int buf) {   // 64 rows x 32 cols, 32B per thread
        int n = threadIdx.x >> 1, h = (threadIdx.x & 1) * 16;
        const __bf16* src = W + (size_t)(nbase + n) * ldw + k0 + h;
        __bf16* dst = Ws + buf * WS_TILE + n * WS_STRIDE + h;
        async_b128(src, dst);
        async_b128(src + 8, dst + 8);
    };

    stage_W(0, 0);
    for (int k0 = 0; k0 < K; k0 += 32) {
        const int buf = (k0 >> 5) & 1;
        wait_async0();          // this wave's current tile is in LDS
        __syncthreads();        // ... and everyone else's part too
        if (k0 + 32 < K) stage_W(k0 + 32, buf ^ 1);   // prefetch next tile

        Frag16 a[2];
#pragma unroll
        for (int mt = 0; mt < 2; ++mt) {
            const __bf16* arow = A + (size_t)(mbase + mt * 16 + nlow) * lda + k0 + kA0;
            a[mt].q[0] = *(const uint4*)(arow);
            a[mt].q[1] = *(const uint4*)(arow + 8);
        }
        const __bf16* wsb = Ws + buf * WS_TILE;
#pragma unroll
        for (int nt = 0; nt < 4; ++nt) {
            Frag16 b;
            const __bf16* wsrc = wsb + (nt * 16 + nlow) * WS_STRIDE + kB0;
            b.q[0] = *(const uint4*)(wsrc);
            b.q[1] = *(const uint4*)(wsrc + 8);
            acc[0][nt] = wmma_bf16(a[0].v, b.v, acc[0][nt]);
            acc[1][nt] = wmma_bf16(a[1].v, b.v, acc[1][nt]);
        }
    }

    // epilogue: C layout -> VGPR r holds rows (r, r+8) across lane halves
#pragma unroll
    for (int mt = 0; mt < 2; ++mt) {
#pragma unroll
        for (int nt = 0; nt < 4; ++nt) {
#pragma unroll
            for (int r = 0; r < 8; ++r) {
                int row = mbase + mt * 16 + r + 8 * hi;
                int col = nbase + nt * 16 + nlow;
                float v = acc[mt][nt][r] + bias[col];
                if (resid) v += resid[(size_t)row * resid_ld + col];
                if (outB) outB[(size_t)row * ldo + col] = (__bf16)v;
                else      outF[(size_t)row * ldo + col] = v;
            }
        }
    }
}

// ---------------------------------------------------------------------------
// RoPE + scatter: qkv[B,N,768] bf16 -> Q,K (rotary applied) in [B*H, N, 64],
// V written PRE-TRANSPOSED: Vt[B*H, 64, N] so flash attention can stage its
// PV B-operand tiles with contiguous async b128 loads (no LDS gather).
// 768-dim layout matches reshape(B,N,H,HD,3): channel = h*192 + d*3 + s
// rotate_half on interleaved pairs: (x0,x1) -> (-x1, x0)
// ---------------------------------------------------------------------------
__global__ void rope_scatter(const __bf16* __restrict__ qkv, const float* __restrict__ kp,
                             __bf16* __restrict__ Qb, __bf16* __restrict__ Kb,
                             __bf16* __restrict__ Vt)
{
    int t = blockIdx.x * 256 + threadIdx.x;          // B*H*N*(HD/2) = 2M threads
    int pair = t & 31;
    int n = (t >> 5) & (N_DIM - 1);
    int h = (t >> 17) & (H_DIM - 1);
    int b = t >> 19;

    size_t base = ((size_t)(b * N_DIM + n)) * 768 + h * 192 + pair * 6;
    float q0 = (float)qkv[base + 0], k0v = (float)qkv[base + 1], v0 = (float)qkv[base + 2];
    float q1 = (float)qkv[base + 3], k1v = (float)qkv[base + 4], v1 = (float)qkv[base + 5];

    size_t kpb = ((size_t)(b * N_DIM + n)) * HD_DIM + pair * 2;
    const size_t SOFF = (size_t)B_DIM * N_DIM * HD_DIM;     // keypoints[1] offset
    float c0 = kp[kpb], c1 = kp[kpb + 1];
    float s0 = kp[SOFF + kpb], s1 = kp[SOFF + kpb + 1];

    float qo0 = q0 * c0 - q1 * s0;
    float qo1 = q1 * c1 + q0 * s1;
    float ko0 = k0v * c0 - k1v * s0;
    float ko1 = k1v * c1 + k0v * s1;

    int bh = b * H_DIM + h;
    size_t ob = ((size_t)(bh * N_DIM + n)) * HD_DIM + pair * 2;
    Qb[ob] = (__bf16)qo0; Qb[ob + 1] = (__bf16)qo1;
    Kb[ob] = (__bf16)ko0; Kb[ob + 1] = (__bf16)ko1;
    size_t vb = ((size_t)bh * HD_DIM + pair * 2) * N_DIM + n;
    Vt[vb]         = (__bf16)v0;
    Vt[vb + N_DIM] = (__bf16)v1;
}

// ---------------------------------------------------------------------------
// Flash attention: grid (N/64, B*H), 128 threads (4 waves).
// Each wave owns 16 query rows; loop over 32-key tiles, K and Vt tiles
// double-buffered in LDS via async b128 loads.
// Per tile per wave: 4 WMMA for S = Q K^T, 4 WMMA for O += P V (K=32).
// P goes through a per-wave LDS scratch to convert C-layout -> A-layout.
// ---------------------------------------------------------------------------
#define KS_STRIDE 72   // 64+8 bf16 -> 144B rows (16B aligned, conflict-free)
#define VT_STRIDE 40   // 32+8 bf16 -> 80B rows
#define PS_STRIDE 40
#define KS_TILE   (32 * KS_STRIDE)
#define VT_TILE   (64 * VT_STRIDE)

__global__ __launch_bounds__(128) void flash_attn(
    const __bf16* __restrict__ Qg, const __bf16* __restrict__ Kg,
    const __bf16* __restrict__ Vtg, __bf16* __restrict__ ctx)
{
    __shared__ __align__(16) __bf16 Ks [2 * KS_TILE];
    __shared__ __align__(16) __bf16 Vts[2 * VT_TILE];
    __shared__ __align__(16) __bf16 Ps [4 * 16 * PS_STRIDE];

    const int lane = threadIdx.x & 31;
    const int w    = threadIdx.x >> 5;
    const int hi   = lane >> 4;
    const int nlow = lane & 15;
    const int bh   = blockIdx.y;
    const int qbase = blockIdx.x * 64 + w * 16;

    const __bf16* Qbh  = Qg  + (size_t)bh * N_DIM * HD_DIM;
    const __bf16* Kbh  = Kg  + (size_t)bh * N_DIM * HD_DIM;
    const __bf16* Vtbh = Vtg + (size_t)bh * HD_DIM * N_DIM;

    // Q fragments (A-layout), 16 rows x 64 feats = 2 fragments, kept in VGPRs
    Frag16 qf[2];
    {
        const __bf16* qrow = Qbh + (size_t)(qbase + nlow) * HD_DIM + hi * 16;
        qf[0].q[0] = *(const uint4*)(qrow);
        qf[0].q[1] = *(const uint4*)(qrow + 8);
        qf[1].q[0] = *(const uint4*)(qrow + 32);
        qf[1].q[1] = *(const uint4*)(qrow + 40);
    }

    v8f o[4] = {};
    float mrow[8], lrow[8];
#pragma unroll
    for (int r = 0; r < 8; ++r) { mrow[r] = -1e30f; lrow[r] = 0.f; }

    __bf16* pw = Ps + w * 16 * PS_STRIDE;     // per-wave P scratch
    const int kB0 = hi * 16;

    auto stage = [&](int j0, int buf) {
        {   // K tile: 32 keys x 64 feats row-major, 32B per thread
            int key = threadIdx.x >> 2, c = (threadIdx.x & 3) * 16;
            const __bf16* src = Kbh + (size_t)(j0 + key) * HD_DIM + c;
            __bf16* dst = Ks + buf * KS_TILE + key * KS_STRIDE + c;
            async_b128(src, dst);
            async_b128(src + 8, dst + 8);
        }
        {   // Vt tile: 64 feats x 32 keys (already transposed in global)
            int f = threadIdx.x >> 1, c = (threadIdx.x & 1) * 16;
            const __bf16* src = Vtbh + (size_t)f * N_DIM + j0 + c;
            __bf16* dst = Vts + buf * VT_TILE + f * VT_STRIDE + c;
            async_b128(src, dst);
            async_b128(src + 8, dst + 8);
        }
    };

    stage(0, 0);
    for (int j0 = 0; j0 < N_DIM; j0 += 32) {
        const int buf = (j0 >> 5) & 1;
        wait_async0();          // this wave's tile parts are in LDS
        __syncthreads();        // everyone's parts visible
        if (j0 + 32 < N_DIM) stage(j0 + 32, buf ^ 1);   // prefetch next tile

        // ---- S = (Q K^T) * scale : two 16x16 n-tiles ----
        v8f s[2] = {};
        const __bf16* ksb = Ks + buf * KS_TILE;
#pragma unroll
        for (int nt = 0; nt < 2; ++nt) {
#pragma unroll
            for (int kk = 0; kk < 2; ++kk) {
                Frag16 bfr;
                const __bf16* ksrc = ksb + (nt * 16 + nlow) * KS_STRIDE + kk * 32 + kB0;
                bfr.q[0] = *(const uint4*)(ksrc);
                bfr.q[1] = *(const uint4*)(ksrc + 8);
                s[nt] = wmma_bf16(qf[kk].v, bfr.v, s[nt]);
            }
        }

        // ---- online softmax (per C-layout row slot; 16-lane half = row) ----
        const float scale = 0.125f;  // HD^-0.5
        float alpha[8];
#pragma unroll
        for (int r = 0; r < 8; ++r) {
            s[0][r] *= scale; s[1][r] *= scale;
            float t = fmaxf(s[0][r], s[1][r]);
            t = fmaxf(t, __shfl_xor(t, 1, 32));
            t = fmaxf(t, __shfl_xor(t, 2, 32));
            t = fmaxf(t, __shfl_xor(t, 4, 32));
            t = fmaxf(t, __shfl_xor(t, 8, 32));
            float mnew = fmaxf(mrow[r], t);
            alpha[r] = __expf(mrow[r] - mnew);
            s[0][r]  = __expf(s[0][r] - mnew);
            s[1][r]  = __expf(s[1][r] - mnew);
            float su = s[0][r] + s[1][r];
            su += __shfl_xor(su, 1, 32);
            su += __shfl_xor(su, 2, 32);
            su += __shfl_xor(su, 4, 32);
            su += __shfl_xor(su, 8, 32);
            lrow[r] = lrow[r] * alpha[r] + su;
            mrow[r] = mnew;
        }
#pragma unroll
        for (int nt = 0; nt < 4; ++nt)
#pragma unroll
            for (int r = 0; r < 8; ++r) o[nt][r] *= alpha[r];

        // ---- P: C-layout -> LDS -> A-layout fragment ----
#pragma unroll
        for (int nt = 0; nt < 2; ++nt)
#pragma unroll
            for (int r = 0; r < 8; ++r)
                pw[(r + 8 * hi) * PS_STRIDE + nt * 16 + nlow] = (__bf16)s[nt][r];
        asm volatile("s_wait_dscnt 0x0" ::: "memory");   // wave-local LDS RAW

        Frag16 pf;
        {
            const __bf16* prow = pw + nlow * PS_STRIDE + hi * 16;
            pf.q[0] = *(const uint4*)(prow);
            pf.q[1] = *(const uint4*)(prow + 8);
        }

        // ---- O += P V : four 16x16 feat tiles, K = 32 keys ----
        const __bf16* vsb = Vts + buf * VT_TILE;
#pragma unroll
        for (int nt = 0; nt < 4; ++nt) {
            Frag16 bfr;
            const __bf16* vsrc = vsb + (nt * 16 + nlow) * VT_STRIDE + kB0;
            bfr.q[0] = *(const uint4*)(vsrc);
            bfr.q[1] = *(const uint4*)(vsrc + 8);
            o[nt] = wmma_bf16(pf.v, bfr.v, o[nt]);
        }
    }

    // ---- normalize and write ctx bf16 [B, N, D] with channel = h*64 + feat ----
    const int h = bh & (H_DIM - 1);
    const int b = bh >> 2;
#pragma unroll
    for (int nt = 0; nt < 4; ++nt) {
#pragma unroll
        for (int r = 0; r < 8; ++r) {
            int row = qbase + r + 8 * hi;
            int col = h * HD_DIM + nt * 16 + nlow;
            ctx[((size_t)(b * N_DIM + row)) * D_DIM + col] = (__bf16)(o[nt][r] / lrow[r]);
        }
    }
}

// ---------------------------------------------------------------------------
// LayerNorm(512) + exact GELU, one wave per row, output bf16
// ---------------------------------------------------------------------------
__global__ __launch_bounds__(256) void ln_gelu(const float* __restrict__ h,
                                               const float* __restrict__ g,
                                               const float* __restrict__ beta,
                                               __bf16* __restrict__ out)
{
    const int lane = threadIdx.x & 31, w = threadIdx.x >> 5;
    const int row = blockIdx.x * 8 + w;
    const float* hr = h + (size_t)row * 512;

    float x[16], s = 0.f, ss = 0.f;
#pragma unroll
    for (int i = 0; i < 16; ++i) {
        x[i] = hr[i * 32 + lane];
        s  += x[i];
        ss += x[i] * x[i];
    }
#pragma unroll
    for (int m = 1; m <= 16; m <<= 1) {
        s  += __shfl_xor(s, m, 32);
        ss += __shfl_xor(ss, m, 32);
    }
    float mu  = s * (1.f / 512.f);
    float var = ss * (1.f / 512.f) - mu * mu;
    float rs  = rsqrtf(var + LN_EPS);
#pragma unroll
    for (int i = 0; i < 16; ++i) {
        int c = i * 32 + lane;
        float v = (x[i] - mu) * rs * g[c] + beta[c];
        v = 0.5f * v * (1.f + erff(v * 0.70710678118f));   // exact GELU
        out[(size_t)row * 512 + c] = (__bf16)v;
    }
}

// ---------------------------------------------------------------------------
// Launch pipeline
// ---------------------------------------------------------------------------
extern "C" void kernel_launch(void* const* d_in, const int* in_sizes, int n_in,
                              void* d_out, int out_size, void* d_ws, size_t ws_size,
                              hipStream_t stream)
{
    const float* descriptors = (const float*)d_in[0];
    const float* keypoints   = (const float*)d_in[1];
    const float* Wqkv_w = (const float*)d_in[2];
    const float* Wqkv_b = (const float*)d_in[3];
    const float* out_w  = (const float*)d_in[4];
    const float* out_b  = (const float*)d_in[5];
    const float* ffn1_w = (const float*)d_in[6];
    const float* ffn1_b = (const float*)d_in[7];
    const float* ln_g   = (const float*)d_in[8];
    const float* ln_b   = (const float*)d_in[9];
    const float* ffn2_w = (const float*)d_in[10];
    const float* ffn2_b = (const float*)d_in[11];
    float* out = (float*)d_out;

    char* wsp = (char*)d_ws;
    auto alloc = [&](size_t bytes) {
        char* p = wsp;
        wsp += (bytes + 255) & ~(size_t)255;
        return p;
    };
    __bf16* wq_bf  = (__bf16*)alloc((size_t)768 * 256 * 2);
    __bf16* wo_bf  = (__bf16*)alloc((size_t)256 * 256 * 2);
    __bf16* w1_bf  = (__bf16*)alloc((size_t)512 * 512 * 2);
    __bf16* w2_bf  = (__bf16*)alloc((size_t)256 * 512 * 2);
    __bf16* hcat   = (__bf16*)alloc((size_t)M_ROWS * 512 * 2);   // [desc | message] bf16
    __bf16* qkv_bf = (__bf16*)alloc((size_t)M_ROWS * 768 * 2);
    __bf16* Qb     = (__bf16*)alloc((size_t)B_DIM * H_DIM * N_DIM * HD_DIM * 2);
    __bf16* Kb     = (__bf16*)alloc((size_t)B_DIM * H_DIM * N_DIM * HD_DIM * 2);
    __bf16* Vtb    = (__bf16*)alloc((size_t)B_DIM * H_DIM * HD_DIM * N_DIM * 2);
    __bf16* ctx_bf = (__bf16*)alloc((size_t)M_ROWS * 256 * 2);
    float*  h_f32  = (float*) alloc((size_t)M_ROWS * 512 * 4);
    __bf16* h2_bf  = (__bf16*)alloc((size_t)M_ROWS * 512 * 2);

    // 1. weight / activation conversions to bf16
    cvt_f32_bf16<<<(768 * 256 + 255) / 256, 256, 0, stream>>>(Wqkv_w, wq_bf, 768 * 256);
    cvt_f32_bf16<<<(256 * 256 + 255) / 256, 256, 0, stream>>>(out_w,  wo_bf, 256 * 256);
    cvt_f32_bf16<<<(512 * 512 + 255) / 256, 256, 0, stream>>>(ffn1_w, w1_bf, 512 * 512);
    cvt_f32_bf16<<<(256 * 512 + 255) / 256, 256, 0, stream>>>(ffn2_w, w2_bf, 256 * 512);
    desc_to_hcat<<<(M_ROWS * 256) / 256, 256, 0, stream>>>(descriptors, hcat);

    // 2. QKV projection: hcat[:, :256] (lda=512) x Wqkv^T -> qkv_bf [16384, 768]
    gemm_wmma<<<dim3(768 / 64, M_ROWS / 128), 128, 0, stream>>>(
        hcat, 512, wq_bf, 256, Wqkv_b, nullptr, 0, nullptr, qkv_bf, 768, 256);

    // 3. RoPE + scatter to per-head Q/K (row-major) and V (transposed)
    rope_scatter<<<(B_DIM * H_DIM * N_DIM * (HD_DIM / 2)) / 256, 256, 0, stream>>>(
        qkv_bf, keypoints, Qb, Kb, Vtb);

    // 4. Flash attention -> ctx bf16 [B, N, 256]
    flash_attn<<<dim3(N_DIM / 64, B_DIM * H_DIM), 128, 0, stream>>>(Qb, Kb, Vtb, ctx_bf);

    // 5. Output projection -> message, written into hcat[:, 256:512]
    gemm_wmma<<<dim3(256 / 64, M_ROWS / 128), 128, 0, stream>>>(
        ctx_bf, 256, wo_bf, 256, out_b, nullptr, 0, nullptr, hcat + 256, 512, 256);

    // 6. FFN1: hcat [16384,512] x ffn1^T -> h_f32 [16384,512]
    gemm_wmma<<<dim3(512 / 64, M_ROWS / 128), 128, 0, stream>>>(
        hcat, 512, w1_bf, 512, ffn1_b, nullptr, 0, h_f32, nullptr, 512, 512);

    // 7. LayerNorm + exact GELU -> h2_bf
    ln_gelu<<<M_ROWS / 8, 256, 0, stream>>>(h_f32, ln_g, ln_b, h2_bf);

    // 8. FFN2 + bias + residual(descriptors) -> d_out f32 [16384,256]
    gemm_wmma<<<dim3(256 / 64, M_ROWS / 128), 128, 0, stream>>>(
        h2_bf, 512, w2_bf, 512, ffn2_b, descriptors, 256, out, nullptr, 256, 512);

    (void)in_sizes; (void)n_in; (void)out_size; (void)ws_size;
}
